// SelfAttention_15015205666857
// MI455X (gfx1250) — compile-verified, matrix operand order checked by reading, not run on hardware
//
#include <hip/hip_runtime.h>

// ---------------------------------------------------------------------------
// Self-attention forward for MI455X (gfx1250), wave32 + WMMA + async LDS.
//   q = x@Wq^T + bq (scaled), k = x@Wk^T + bk, v = x@Wv^T + bv (stored v^T)
//   attn = softmax(q@k^T)          -> d_out[0 .. B*S*S)
//   out  = attn @ v                -> d_out[B*S*S .. )
// GEMMs: 128x128 block tile, 8 waves (2x4), wave tile 64x32 (4x2 WMMA tiles),
// K-panel = 32, double-buffered LDS staged with GLOBAL_LOAD_ASYNC_TO_LDS_B128.
// ---------------------------------------------------------------------------

typedef __attribute__((ext_vector_type(16))) _Float16 v16h;
typedef __attribute__((ext_vector_type(8)))  _Float16 v8h;
typedef __attribute__((ext_vector_type(8)))  float    v8f;
typedef __attribute__((ext_vector_type(4)))  float    v4f;
typedef __attribute__((ext_vector_type(4)))  int      v4i;

#define DM    1024
#define SEQ   2048
#define BATCH 4
#define MTOT  (BATCH * SEQ)     // 8192 flattened rows
#define QK_SCALE 0.03125f       // 1/sqrt(1024)

#define LDSPITCH 40             // padded row pitch (elements) - kills bank conflicts
#define TILE_ELEMS (128 * LDSPITCH)   // 5120

// ---------------- workspace layout (element offsets, _Float16) -------------
#define WS_XB   0                                    // [8192][1024] x in f16
#define WS_WQ   (WS_XB + MTOT * DM)                  // [1024][1024]
#define WS_WK   (WS_WQ + DM * DM)
#define WS_WV   (WS_WK + DM * DM)
#define WS_QB   (WS_WV + DM * DM)                    // [8192][1024] scaled q
#define WS_KB   (WS_QB + MTOT * DM)                  // [8192][1024]
#define WS_VT   (WS_KB + MTOT * DM)                  // [4][1024][2048] v^T
// total = 36,700,160 halves = ~70 MB

// ---------------------------------------------------------------------------
// Async global->LDS 16-byte copy (gfx1250 GLOBAL_LOAD_ASYNC_TO_LDS_B128,
// tracked by ASYNCcnt).  Builtin signature (from compiler diagnostic):
//   (v4i addrspace(1)* src, v4i addrspace(3)* dst, imm offset, imm cpol)
// Fallback: synchronous reg round-trip.
// ---------------------------------------------------------------------------
#if __has_builtin(__builtin_amdgcn_global_load_async_to_lds_b128)
#define HAVE_ASYNC 1
#else
#define HAVE_ASYNC 0
#endif

__device__ __forceinline__ void copy16(void* lds, const void* g) {
#if HAVE_ASYNC
  __builtin_amdgcn_global_load_async_to_lds_b128(
      (__attribute__((address_space(1))) v4i*)g,
      (__attribute__((address_space(3))) v4i*)lds,
      /*offset=*/0, /*cpol=*/0);
#else
  *(v4i*)lds = *(const v4i*)g;
#endif
}

__device__ __forceinline__ void wait_stage() {
#if HAVE_ASYNC
#if __has_builtin(__builtin_amdgcn_s_wait_asynccnt)
  __builtin_amdgcn_s_wait_asynccnt(0);
#else
  asm volatile("s_wait_asynccnt 0x0" ::: "memory");
#endif
#endif
  __syncthreads();
}

// Stage a 128-row x 32-elem K-panel into LDS (padded pitch LDSPITCH).
// f16 panel: 512 16B chunks, 256 threads x 2.
__device__ __forceinline__ void stage_f16(_Float16* lds, const _Float16* g,
                                          int ldg, int k0) {
  const int t = threadIdx.x;
#pragma unroll
  for (int c = 0; c < 2; ++c) {
    const int chunk = t + c * 256;
    const int row = chunk >> 2, seg = chunk & 3;       // 4 x 8-half segments
    copy16(lds + row * LDSPITCH + seg * 8,
           g + (long)row * ldg + k0 + seg * 8);
  }
}

// f32 panel: 1024 16B chunks, 256 threads x 4.
__device__ __forceinline__ void stage_f32(float* lds, const float* g,
                                          int ldg, int k0) {
  const int t = threadIdx.x;
#pragma unroll
  for (int c = 0; c < 4; ++c) {
    const int chunk = t + c * 256;
    const int row = chunk >> 3, seg = chunk & 7;       // 8 x 4-float segments
    copy16(lds + row * LDSPITCH + seg * 4,
           g + (long)row * ldg + k0 + seg * 4);
  }
}

// ---------------------------------------------------------------------------
// Fragment loads from LDS (ISA 16-bit A/B striping, §7.12.2):
// lane l -> row (r0 + (l&15)); K-halves {koff..koff+7, koff+16..koff+23},
// koff = (l>>4)*8.  Two ds_load_b128 per fragment.
// ---------------------------------------------------------------------------
__device__ __forceinline__ v16h lds_frag_h(const _Float16* base, int r0) {
  const int lane = threadIdx.x & 31;
  const _Float16* p = base + (r0 + (lane & 15)) * LDSPITCH + ((lane >> 4) << 3);
  v8h lo = *(const v8h*)p;
  v8h hi = *(const v8h*)(p + 16);
  v16h r;
#pragma unroll
  for (int i = 0; i < 8; ++i) { r[i] = lo[i]; r[i + 8] = hi[i]; }
  return r;
}

__device__ __forceinline__ v16h lds_frag_f(const float* base, int r0) {
  const int lane = threadIdx.x & 31;
  const float* p = base + (r0 + (lane & 15)) * LDSPITCH + ((lane >> 4) << 3);
  v4f a0 = *(const v4f*)p;
  v4f a1 = *(const v4f*)(p + 4);
  v4f b0 = *(const v4f*)(p + 16);
  v4f b1 = *(const v4f*)(p + 20);
  v16h r;
#pragma unroll
  for (int i = 0; i < 4; ++i) {
    r[i]      = (_Float16)a0[i];
    r[i + 4]  = (_Float16)a1[i];
    r[i + 8]  = (_Float16)b0[i];
    r[i + 12] = (_Float16)b1[i];
  }
  return r;
}

// ---------------------------------------------------------------------------
// Double-buffered main loops.  Wave tile 64x32 at (m0w, n0w); 8 WMMA per step.
// ---------------------------------------------------------------------------
__device__ __forceinline__ void gemm_loop_f16(
    const _Float16* __restrict__ Ag, int lda,
    const _Float16* __restrict__ Bg, int ldb, int K,
    _Float16 (*sA)[TILE_ELEMS], _Float16 (*sB)[TILE_ELEMS],
    int m0w, int n0w, v8f acc[4][2]) {
  stage_f16(sA[0], Ag, lda, 0);
  stage_f16(sB[0], Bg, ldb, 0);
  wait_stage();
  const int nIter = K / 32;
  for (int it = 0; it < nIter; ++it) {
    const int cur = it & 1;
    if (it + 1 < nIter) {
      stage_f16(sA[cur ^ 1], Ag, lda, (it + 1) * 32);
      stage_f16(sB[cur ^ 1], Bg, ldb, (it + 1) * 32);
    }
    v16h af[4], bf[2];
#pragma unroll
    for (int i = 0; i < 4; ++i) af[i] = lds_frag_h(sA[cur], m0w + 16 * i);
#pragma unroll
    for (int j = 0; j < 2; ++j) bf[j] = lds_frag_h(sB[cur], n0w + 16 * j);
#pragma unroll
    for (int i = 0; i < 4; ++i)
#pragma unroll
      for (int j = 0; j < 2; ++j)
        acc[i][j] = __builtin_amdgcn_wmma_f32_16x16x32_f16(
            false, af[i], false, bf[j], (short)0, acc[i][j], false, false);
    if (it + 1 < nIter) wait_stage();
  }
}

__device__ __forceinline__ void gemm_loop_f32A(
    const float* __restrict__ Ag, int lda,
    const _Float16* __restrict__ Bg, int ldb, int K,
    float (*sA)[TILE_ELEMS], _Float16 (*sB)[TILE_ELEMS],
    int m0w, int n0w, v8f acc[4][2]) {
  stage_f32(sA[0], Ag, lda, 0);
  stage_f16(sB[0], Bg, ldb, 0);
  wait_stage();
  const int nIter = K / 32;
  for (int it = 0; it < nIter; ++it) {
    const int cur = it & 1;
    if (it + 1 < nIter) {
      stage_f32(sA[cur ^ 1], Ag, lda, (it + 1) * 32);
      stage_f16(sB[cur ^ 1], Bg, ldb, (it + 1) * 32);
    }
    v16h af[4], bf[2];
#pragma unroll
    for (int i = 0; i < 4; ++i) af[i] = lds_frag_f(sA[cur], m0w + 16 * i);
#pragma unroll
    for (int j = 0; j < 2; ++j) bf[j] = lds_frag_h(sB[cur], n0w + 16 * j);
#pragma unroll
    for (int i = 0; i < 4; ++i)
#pragma unroll
      for (int j = 0; j < 2; ++j)
        acc[i][j] = __builtin_amdgcn_wmma_f32_16x16x32_f16(
            false, af[i], false, bf[j], (short)0, acc[i][j], false, false);
    if (it + 1 < nIter) wait_stage();
  }
}

// ---------------------------------------------------------------------------
// f32 -> f16 pack
// ---------------------------------------------------------------------------
__global__ void cvt_f32_to_f16(const float* __restrict__ src,
                               _Float16* __restrict__ dst, int n) {
  int i = blockIdx.x * blockDim.x + threadIdx.x;
  if (i < n) dst[i] = (_Float16)src[i];
}

// ---------------------------------------------------------------------------
// Fused QKV projection.  grid = (MTOT/128, DM/128, 3), block = 256.
// ---------------------------------------------------------------------------
__global__ void qkv_gemm(const _Float16* __restrict__ xb,
                         const _Float16* __restrict__ wq16,
                         const _Float16* __restrict__ wk16,
                         const _Float16* __restrict__ wv16,
                         const float* __restrict__ bq,
                         const float* __restrict__ bk,
                         const float* __restrict__ bv,
                         _Float16* __restrict__ qb,
                         _Float16* __restrict__ kb,
                         _Float16* __restrict__ vT) {
  __shared__ _Float16 sA[2][TILE_ELEMS];
  __shared__ _Float16 sB[2][TILE_ELEMS];
  const int z = blockIdx.z;
  const int blockM = blockIdx.x * 128;
  const int blockN = blockIdx.y * 128;
  const int wid = threadIdx.x >> 5, lane = threadIdx.x & 31;
  const int m0w = (wid >> 2) * 64, n0w = (wid & 3) * 32;

  const _Float16* W    = (z == 0) ? wq16 : (z == 1) ? wk16 : wv16;
  const float*    bias = (z == 0) ? bq   : (z == 1) ? bk   : bv;

  v8f acc[4][2];
#pragma unroll
  for (int i = 0; i < 4; ++i)
#pragma unroll
    for (int j = 0; j < 2; ++j) acc[i][j] = (v8f){};

  gemm_loop_f16(xb + (long)blockM * DM, DM, W + (long)blockN * DM, DM, DM,
                sA, sB, m0w, n0w, acc);

  const int nl = lane & 15, sel = lane >> 4;
  if (z < 2) {
    _Float16* outp = (z == 0) ? qb : kb;
    const float scl = (z == 0) ? QK_SCALE : 1.0f;
#pragma unroll
    for (int i = 0; i < 4; ++i)
#pragma unroll
      for (int j = 0; j < 2; ++j) {
        const int ncol = blockN + n0w + j * 16 + nl;
        const float bvl = bias[ncol];
#pragma unroll
        for (int r = 0; r < 8; ++r) {
          const int m = blockM + m0w + i * 16 + r + 8 * sel;
          outp[(long)m * DM + ncol] = (_Float16)((acc[i][j][r] + bvl) * scl);
        }
      }
  } else {
    // v stored transposed: vT[b][e][s]; per-lane contiguous along s
    const int b  = blockM >> 11;
    const int s0 = blockM & (SEQ - 1);
    _Float16* outp = vT + (long)b * DM * SEQ;
#pragma unroll
    for (int i = 0; i < 4; ++i)
#pragma unroll
      for (int j = 0; j < 2; ++j) {
        const int e = blockN + n0w + j * 16 + nl;
        const float bvl = bias[e];
        _Float16* row = outp + (long)e * SEQ + s0 + m0w + i * 16 + 8 * sel;
#pragma unroll
        for (int r = 0; r < 8; ++r) row[r] = (_Float16)(acc[i][j][r] + bvl);
      }
  }
}

// ---------------------------------------------------------------------------
// logits = q @ k^T (scale pre-folded into q).
// grid = (SEQ/128, SEQ/128, BATCH), block = 256.
// ---------------------------------------------------------------------------
__global__ void logits_gemm(const _Float16* __restrict__ qb,
                            const _Float16* __restrict__ kb,
                            float* __restrict__ attn) {
  __shared__ _Float16 sA[2][TILE_ELEMS];
  __shared__ _Float16 sB[2][TILE_ELEMS];
  const int b = blockIdx.z;
  const int blockM = blockIdx.x * 128;
  const int blockN = blockIdx.y * 128;
  const int wid = threadIdx.x >> 5, lane = threadIdx.x & 31;
  const int m0w = (wid >> 2) * 64, n0w = (wid & 3) * 32;

  const _Float16* A  = qb + (long)b * SEQ * DM + (long)blockM * DM;
  const _Float16* Bm = kb + (long)b * SEQ * DM + (long)blockN * DM;

  v8f acc[4][2];
#pragma unroll
  for (int i = 0; i < 4; ++i)
#pragma unroll
    for (int j = 0; j < 2; ++j) acc[i][j] = (v8f){};

  gemm_loop_f16(A, DM, Bm, DM, DM, sA, sB, m0w, n0w, acc);

  float* out = attn + (long)b * SEQ * SEQ;
  const int nl = lane & 15, sel = lane >> 4;
#pragma unroll
  for (int i = 0; i < 4; ++i)
#pragma unroll
    for (int j = 0; j < 2; ++j) {
      const int ncol = blockN + n0w + j * 16 + nl;
#pragma unroll
      for (int r = 0; r < 8; ++r) {
        const int m = blockM + m0w + i * 16 + r + 8 * sel;
        out[(long)m * SEQ + ncol] = acc[i][j][r];
      }
    }
}

// ---------------------------------------------------------------------------
// In-place row softmax over 2048 elements.  grid = BATCH*SEQ, block = 256.
// ---------------------------------------------------------------------------
__global__ void softmax_rows(float* __restrict__ attn) {
  __shared__ float red[256];
  float* p = attn + (long)blockIdx.x * SEQ;
  const int t = threadIdx.x;

  float v[8];
  float mx = -3.4e38f;
#pragma unroll
  for (int i = 0; i < 8; ++i) { v[i] = p[t + i * 256]; mx = fmaxf(mx, v[i]); }
  red[t] = mx; __syncthreads();
  for (int s = 128; s > 0; s >>= 1) {
    if (t < s) red[t] = fmaxf(red[t], red[t + s]);
    __syncthreads();
  }
  mx = red[0]; __syncthreads();

  float sum = 0.f;
#pragma unroll
  for (int i = 0; i < 8; ++i) { v[i] = __expf(v[i] - mx); sum += v[i]; }
  red[t] = sum; __syncthreads();
  for (int s = 128; s > 0; s >>= 1) {
    if (t < s) red[t] += red[t + s];
    __syncthreads();
  }
  const float inv = 1.0f / red[0];
#pragma unroll
  for (int i = 0; i < 8; ++i) p[t + i * 256] = v[i] * inv;
}

// ---------------------------------------------------------------------------
// out = attn @ v using vT.  A (attn) is f32: staged to LDS, cvt on frag load.
// grid = (SEQ/128, DM/128, BATCH), block = 256.
// ---------------------------------------------------------------------------
__global__ void out_gemm(const float* __restrict__ attn,
                         const _Float16* __restrict__ vT,
                         float* __restrict__ out) {
  __shared__ float    sA[2][TILE_ELEMS];   // 40 KB
  __shared__ _Float16 sB[2][TILE_ELEMS];   // 20 KB
  const int b = blockIdx.z;
  const int blockM = blockIdx.x * 128;
  const int blockN = blockIdx.y * 128;
  const int wid = threadIdx.x >> 5, lane = threadIdx.x & 31;
  const int m0w = (wid >> 2) * 64, n0w = (wid & 3) * 32;

  const float*    A  = attn + (long)b * SEQ * SEQ + (long)blockM * SEQ;
  const _Float16* Bm = vT + (long)b * DM * SEQ + (long)blockN * SEQ;

  v8f acc[4][2];
#pragma unroll
  for (int i = 0; i < 4; ++i)
#pragma unroll
    for (int j = 0; j < 2; ++j) acc[i][j] = (v8f){};

  gemm_loop_f32A(A, SEQ, Bm, SEQ, SEQ, sA, sB, m0w, n0w, acc);

  float* O = out + (long)b * SEQ * DM;
  const int nl = lane & 15, sel = lane >> 4;
#pragma unroll
  for (int i = 0; i < 4; ++i)
#pragma unroll
    for (int j = 0; j < 2; ++j) {
      const int ncol = blockN + n0w + j * 16 + nl;
#pragma unroll
      for (int r = 0; r < 8; ++r) {
        const int m = blockM + m0w + i * 16 + r + 8 * sel;
        O[(long)m * DM + ncol] = acc[i][j][r];
      }
    }
}

// ---------------------------------------------------------------------------
extern "C" void kernel_launch(void* const* d_in, const int* in_sizes, int n_in,
                              void* d_out, int out_size, void* d_ws, size_t ws_size,
                              hipStream_t stream) {
  const float* x  = (const float*)d_in[0];
  const float* wq = (const float*)d_in[1];
  const float* bq = (const float*)d_in[2];
  const float* wk = (const float*)d_in[3];
  const float* bk = (const float*)d_in[4];
  const float* wv = (const float*)d_in[5];
  const float* bv = (const float*)d_in[6];

  float* attn = (float*)d_out;                       // [4][2048][2048]
  float* out  = attn + (long)BATCH * SEQ * SEQ;      // [4][2048][1024]

  _Float16* ws   = (_Float16*)d_ws;
  _Float16* xb   = ws + WS_XB;
  _Float16* wq16 = ws + WS_WQ;
  _Float16* wk16 = ws + WS_WK;
  _Float16* wv16 = ws + WS_WV;
  _Float16* qb   = ws + WS_QB;
  _Float16* kb   = ws + WS_KB;
  _Float16* vT   = ws + WS_VT;

  // 1) pack to f16
  const int nX = MTOT * DM, nW = DM * DM;
  cvt_f32_to_f16<<<(nX + 255) / 256, 256, 0, stream>>>(x, xb, nX);
  cvt_f32_to_f16<<<(nW + 255) / 256, 256, 0, stream>>>(wq, wq16, nW);
  cvt_f32_to_f16<<<(nW + 255) / 256, 256, 0, stream>>>(wk, wk16, nW);
  cvt_f32_to_f16<<<(nW + 255) / 256, 256, 0, stream>>>(wv, wv16, nW);

  // 2) fused QKV projection
  dim3 g1(MTOT / 128, DM / 128, 3);
  qkv_gemm<<<g1, 256, 0, stream>>>(xb, wq16, wk16, wv16, bq, bk, bv, qb, kb, vT);

  // 3) logits
  dim3 g2(SEQ / 128, SEQ / 128, BATCH);
  logits_gemm<<<g2, 256, 0, stream>>>(qb, kb, attn);

  // 4) softmax (in place on d_out attn region)
  softmax_rows<<<BATCH * SEQ, 256, 0, stream>>>(attn);

  // 5) out = attn @ v
  dim3 g3(SEQ / 128, DM / 128, BATCH);
  out_gemm<<<g3, 256, 0, stream>>>(attn, vT, out);
}
